// SMKMoE_33097017983631
// MI455X (gfx1250) — matrix operand order repaired and checked by reading, not compiled, not run on hardware
//
#include <hip/hip_runtime.h>
#include <hip/hip_bf16.h>
#include <math.h>

// ---------------------------------------------------------------------------
// Problem constants (from reference)
// ---------------------------------------------------------------------------
#define NTOK   8192      // B*T
#define CH     1024      // HIDDEN
#define IN     256       // INTER
#define NE     16        // N_EXPERTS
#define NEG_INF (-1000000000.0f)

typedef __attribute__((ext_vector_type(16))) __bf16 v16bf;
typedef __attribute__((ext_vector_type(8)))  __bf16 v8bf;
typedef __attribute__((ext_vector_type(8)))  float  v8f;

union Frag16 { v16bf v; v8bf h[2]; };

// LDS staging geometry shared by both GEMMs:
//   chunk = 64 rows x 256 bf16 (row = 512 bytes), padded row stride = 264
//   elements (528 B = 132 DWORDs; 132 mod 64 = 4 banks -> conflict-free frags)
#define STG_ROWS   64
#define STG_K      256
#define STG_STEPS  (STG_K / 32)        // 8 k-steps per chunk
#define STG_PITCH  264                 // elements (512 B data + 16 B pad)
#define STG_ELEMS  (STG_ROWS * STG_PITCH)
#define PAD_INTERVAL_ENC 6u            // 512-byte interval
#define PAD_AMOUNT_ENC   3u            // 16 bytes (4 DWORDs)

#ifndef __has_builtin
#define __has_builtin(x) 0
#endif
#if __has_builtin(__builtin_amdgcn_tensor_load_to_lds)
#define HAVE_TDM 1
#else
#define HAVE_TDM 0
#endif

#if HAVE_TDM
typedef unsigned int tdm_v4u __attribute__((ext_vector_type(4)));
typedef int          tdm_v8i __attribute__((ext_vector_type(8)));
typedef int          tdm_v4i __attribute__((ext_vector_type(4)));

// Issue a 2D tile DMA: rows x rowElems bf16 tile, global row stride
// strideElems, into LDS at byte offset lds_off with per-row padding.
// Descriptor packing per CDNA5 ISA §8.3/8.4 (D# group0/group1).
__device__ __forceinline__ void tdm_issue_2d(unsigned lds_off, const void* gsrc,
                                             unsigned rowElems, unsigned rows,
                                             unsigned long long strideElems) {
    unsigned long long ga = (unsigned long long)(size_t)gsrc;
    tdm_v4u g0;
    g0[0] = 1u;                                        // count=1, user mode
    g0[1] = lds_off;                                   // lds_addr
    g0[2] = (unsigned)(ga & 0xFFFFFFFFu);              // global_addr[31:0]
    g0[3] = (unsigned)((ga >> 32) & 0x01FFFFFFu)       // global_addr[56:32]
          | (2u << 30);                                // type = 2 (image)
    tdm_v8i g1;
    g1[0] = (int)((1u << 16)                           // data_size = 2 bytes
          | (1u << 20)                                 // pad_enable
          | (PAD_INTERVAL_ENC << 22)
          | (PAD_AMOUNT_ENC << 25));
    g1[1] = (int)((rowElems & 0xFFFFu) << 16);                              // td0[15:0]
    g1[2] = (int)(((rowElems >> 16) & 0xFFFFu) | ((rows & 0xFFFFu) << 16)); // td0 hi | td1 lo
    g1[3] = (int)(((rows >> 16) & 0xFFFFu) | ((rowElems & 0xFFFFu) << 16)); // td1 hi | tile_dim0
    g1[4] = (int)(rows & 0xFFFFu);                     // tile_dim1 ; tile_dim2=0
    g1[5] = (int)(strideElems & 0xFFFFFFFFull);        // tensor_dim0_stride lo
    g1[6] = (int)((strideElems >> 32) & 0xFFFFull);    // stride hi ; dim1_stride=0
    g1[7] = 0;
    tdm_v4i z = {0, 0, 0, 0};
#if defined(__clang_major__) && (__clang_major__ >= 23)
    tdm_v8i z8 = {0, 0, 0, 0, 0, 0, 0, 0};
    __builtin_amdgcn_tensor_load_to_lds(g0, g1, z, z, z8, 0);
#else
    __builtin_amdgcn_tensor_load_to_lds(g0, g1, z, z, 0);
#endif
}
#endif  // HAVE_TDM

__device__ __forceinline__ unsigned lds_offset_of(const void* p) {
    // Low 32 bits of a generic pointer into LDS = byte offset in wave's LDS.
    return (unsigned)(size_t)p;
}

// ---------------------------------------------------------------------------
// Pipelined 32x64 x (K=256) chunk MMA (2 M-tiles x 4 N-tiles, 8 accumulators).
//   Arow0/Arow1 : global A bases for the two 16-row M-tiles (row = lane&15)
//   Bbase       : LDS staged 64 x 256 panel (padded pitch)
// Each B fragment read from LDS feeds TWO back-to-back WMMAs; each A fragment
// feeds four -> dependency waits are amortized over 2-4 matrix ops.
// ---------------------------------------------------------------------------
__device__ __forceinline__ void chunk_mma2(const __bf16* __restrict__ Arow0,
                                           const __bf16* __restrict__ Arow1,
                                           int kbase,
                                           const __bf16* __restrict__ Bbase,
                                           int halfsel, int col, v8f acc[2][4]) {
#pragma unroll
    for (int ks = 0; ks < STG_STEPS; ++ks) {
        const int ka = kbase + ks * 32;
        Frag16 a[2];
        a[0].h[0] = *(const v8bf*)(Arow0 + ka + halfsel * 8);
        a[0].h[1] = *(const v8bf*)(Arow0 + ka + 16 + halfsel * 8);
        a[1].h[0] = *(const v8bf*)(Arow1 + ka + halfsel * 8);
        a[1].h[1] = *(const v8bf*)(Arow1 + ka + 16 + halfsel * 8);
        const int kk = ks * 32 + halfsel * 8;
        Frag16 b[4];
#pragma unroll
        for (int j = 0; j < 4; ++j) {
            const __bf16* bp = Bbase + (j * 16 + col) * STG_PITCH + kk;
            b[j].h[0] = *(const v8bf*)bp;
            b[j].h[1] = *(const v8bf*)(bp + 16);
        }
#pragma unroll
        for (int j = 0; j < 4; ++j) {
#pragma unroll
            for (int mi = 0; mi < 2; ++mi) {
                acc[mi][j] = __builtin_amdgcn_wmma_f32_16x16x32_bf16(
                    false, a[mi].v, false, b[j].v, (short)0, acc[mi][j], false, false);
            }
        }
    }
}

// ---------------------------------------------------------------------------
// Elementwise conversions / transposes (bf16 staging in workspace)
// ---------------------------------------------------------------------------
__global__ void moe_conv_x(const float* __restrict__ x, __bf16* __restrict__ xb) {
    size_t idx = (size_t)blockIdx.x * 256 + threadIdx.x;   // N*C elements
    xb[idx] = (__bf16)x[idx];
}

// w1 [E, C, I] -> w1t [E, I, C]
__global__ void moe_conv_w1(const float* __restrict__ w1, __bf16* __restrict__ w1t) {
    size_t idx = (size_t)blockIdx.x * 256 + threadIdx.x;   // E*I*C (out idx)
    int c = (int)(idx & (CH - 1));
    size_t t = idx >> 10;
    int i = (int)(t & (IN - 1));
    int e = (int)(t >> 8);
    w1t[idx] = (__bf16)w1[((size_t)e * CH + c) * IN + i];
}

// w2 [E, I, C] -> w2t [E, C, I]
__global__ void moe_conv_w2(const float* __restrict__ w2, __bf16* __restrict__ w2t) {
    size_t idx = (size_t)blockIdx.x * 256 + threadIdx.x;   // E*C*I (out idx)
    int i = (int)(idx & (IN - 1));
    size_t t = idx >> 8;
    int c = (int)(t & (CH - 1));
    int e = (int)(t >> 10);
    w2t[idx] = (__bf16)w2[((size_t)e * IN + i) * CH + c];
}

// ---------------------------------------------------------------------------
// Gating: one wave32 per token.
// ---------------------------------------------------------------------------
__global__ void moe_gate(const float* __restrict__ x, const float* __restrict__ sim,
                         const float* __restrict__ thr,
                         float* __restrict__ scores_out, int* __restrict__ k_out,
                         float* __restrict__ rw_out) {
    const int token = blockIdx.x * (blockDim.x >> 5) + (threadIdx.x >> 5);
    const int lane  = threadIdx.x & 31;
    if (token >= NTOK) return;

    const float* xr = x + (size_t)token * CH;
    float ss = 0.0f;
    float s[NE], sn[NE];
#pragma unroll
    for (int e = 0; e < NE; ++e) { s[e] = 0.0f; sn[e] = 0.0f; }

    for (int c = lane; c < CH; c += 32) {
        float v = xr[c];
        ss += v * v;
#pragma unroll
        for (int e = 0; e < NE; ++e) {
            float w = sim[c * NE + e];
            s[e]  += v * w;
            sn[e] += w * w;
        }
    }
#pragma unroll
    for (int off = 16; off > 0; off >>= 1) {
        ss += __shfl_xor(ss, off, 32);
#pragma unroll
        for (int e = 0; e < NE; ++e) {
            s[e]  += __shfl_xor(s[e],  off, 32);
            sn[e] += __shfl_xor(sn[e], off, 32);
        }
    }

    const float xn = fmaxf(sqrtf(ss), 1e-12f);
    const float t  = thr[0];

    float sc[NE], ex[NE];
    int   k  = 0;
    float mx = NEG_INF;
#pragma unroll
    for (int e = 0; e < NE; ++e) {
        sc[e] = s[e] / (xn * fmaxf(sqrtf(sn[e]), 1e-12f));
        bool act = sc[e] > t;
        k += act ? 1 : 0;
        float m = act ? sc[e] : NEG_INF;
        ex[e] = m;
        mx = fmaxf(mx, m);
    }
    float den = 0.0f;
#pragma unroll
    for (int e = 0; e < NE; ++e) {
        ex[e] = __expf(ex[e] - mx);
        den += ex[e];
    }
    if (lane == 0) {
        float inv = 1.0f / den;
#pragma unroll
        for (int e = 0; e < NE; ++e) {
            scores_out[(size_t)token * NE + e] = sc[e];
            rw_out[(size_t)token * NE + e]     = ex[e] * inv;
        }
        k_out[token] = k;
    }
}

// ---------------------------------------------------------------------------
// GEMM1: H[e][n][i] = bf16( rw[n,e] * gelu( sum_c xb[n,c] * w1t[e,i,c] ) )
// Block = 8 waves; wave = 32 token rows x 64 inter cols for one (expert, col
// group). B panel TDM-staged to LDS in 4 double-buffered chunks of K=256.
// ---------------------------------------------------------------------------
__global__ void moe_gemm1(const __bf16* __restrict__ xb, const __bf16* __restrict__ w1t,
                          const float* __restrict__ rw, __bf16* __restrict__ Hb) {
    __shared__ __bf16 smem[2][STG_ELEMS];

    const int bg = blockIdx.x & 31;          // token group (32 groups of 8x32 rows)
    const int tn = (blockIdx.x >> 5) & 3;    // 64-col group within IN
    const int e  = blockIdx.x >> 7;          // expert
    const int w       = threadIdx.x >> 5;
    const int lane    = threadIdx.x & 31;
    const int halfsel = lane >> 4;
    const int col     = lane & 15;
    const int mbase   = (bg * 8 + w) * 32;   // first of this wave's 32 token rows

    const __bf16* Bpanel = w1t + ((size_t)e * IN + tn * 64) * CH;   // 64 rows x CH
    const bool leader = (threadIdx.x < 32);

#if HAVE_TDM
    if (leader) tdm_issue_2d(lds_offset_of(&smem[0][0]), Bpanel, STG_K, STG_ROWS, CH);
#endif

    const __bf16* Arow0 = xb + (size_t)(mbase + col) * CH;
    const __bf16* Arow1 = xb + (size_t)(mbase + 16 + col) * CH;
    v8f acc[2][4] = {};

#pragma unroll
    for (int c = 0; c < 4; ++c) {
#if HAVE_TDM
        if (leader) {
            if (c + 1 < 4) {
                tdm_issue_2d(lds_offset_of(&smem[(c + 1) & 1][0]),
                             Bpanel + (c + 1) * STG_K, STG_K, STG_ROWS, CH);
                __builtin_amdgcn_s_wait_tensorcnt(1);
            } else {
                __builtin_amdgcn_s_wait_tensorcnt(0);
            }
        }
#else
        for (int u = threadIdx.x; u < STG_ROWS * (STG_K / 8); u += 256) {
            int r = u >> 5, k8 = (u & 31) * 8;
            *(v8bf*)&smem[c & 1][r * STG_PITCH + k8] =
                *(const v8bf*)(Bpanel + (size_t)r * CH + c * STG_K + k8);
        }
#endif
        __syncthreads();
        chunk_mma2(Arow0, Arow1, c * STG_K, &smem[c & 1][0], halfsel, col, acc);
        __syncthreads();
    }

#pragma unroll
    for (int mi = 0; mi < 2; ++mi) {
#pragma unroll
        for (int j = 0; j < 4; ++j) {
#pragma unroll
            for (int r = 0; r < 8; ++r) {
                const int m = mbase + mi * 16 + r + halfsel * 8;
                float g  = acc[mi][j][r];
                float ge = 0.5f * g * (1.0f + erff(g * 0.70710678118654752f)); // exact gelu
                float o  = ge * rw[(size_t)m * NE + e];
                Hb[((size_t)e * NTOK + m) * IN + tn * 64 + j * 16 + col] = (__bf16)o;
            }
        }
    }
}

// ---------------------------------------------------------------------------
// GEMM2: final[n][c] = sum_e sum_i Hb[e][n][i] * w2t[e][c][i]
// Block = 8 waves; wave = 32 token rows x 64 cols of CH; loops all 16 experts
// with TDM double-buffered B panels; 1024 accumulating WMMAs per wave.
// ---------------------------------------------------------------------------
__global__ void moe_gemm2(const __bf16* __restrict__ Hb, const __bf16* __restrict__ w2t,
                          float* __restrict__ out) {
    __shared__ __bf16 smem[2][STG_ELEMS];

    const int bg = blockIdx.x & 31;          // token group
    const int tn = blockIdx.x >> 5;          // 64-col group within CH (16 groups)
    const int w       = threadIdx.x >> 5;
    const int lane    = threadIdx.x & 31;
    const int halfsel = lane >> 4;
    const int col     = lane & 15;
    const int mbase   = (bg * 8 + w) * 32;
    const bool leader = (threadIdx.x < 32);

#if HAVE_TDM
    if (leader) tdm_issue_2d(lds_offset_of(&smem[0][0]),
                             w2t + ((size_t)0 * CH + tn * 64) * IN, STG_K, STG_ROWS, IN);
#endif

    v8f acc[2][4] = {};
    for (int e = 0; e < NE; ++e) {
#if HAVE_TDM
        if (leader) {
            if (e + 1 < NE) {
                tdm_issue_2d(lds_offset_of(&smem[(e + 1) & 1][0]),
                             w2t + ((size_t)(e + 1) * CH + tn * 64) * IN,
                             STG_K, STG_ROWS, IN);
                __builtin_amdgcn_s_wait_tensorcnt(1);
            } else {
                __builtin_amdgcn_s_wait_tensorcnt(0);
            }
        }
#else
        for (int u = threadIdx.x; u < STG_ROWS * (STG_K / 8); u += 256) {
            int r = u >> 5, k8 = (u & 31) * 8;
            *(v8bf*)&smem[e & 1][r * STG_PITCH + k8] =
                *(const v8bf*)(w2t + ((size_t)e * CH + tn * 64 + r) * IN + k8);
        }
#endif
        __syncthreads();
        const __bf16* Arow0 = Hb + ((size_t)e * NTOK + mbase + col) * IN;
        const __bf16* Arow1 = Hb + ((size_t)e * NTOK + mbase + 16 + col) * IN;
        chunk_mma2(Arow0, Arow1, 0, &smem[e & 1][0], halfsel, col, acc);
        __syncthreads();
    }

#pragma unroll
    for (int mi = 0; mi < 2; ++mi) {
#pragma unroll
        for (int j = 0; j < 4; ++j) {
#pragma unroll
            for (int r = 0; r < 8; ++r) {
                const int m = mbase + mi * 16 + r + halfsel * 8;
                out[(size_t)m * CH + tn * 64 + j * 16 + col] = acc[mi][j][r];
            }
        }
    }
}

// ---------------------------------------------------------------------------
// Host launcher
// ---------------------------------------------------------------------------
extern "C" void kernel_launch(void* const* d_in, const int* in_sizes, int n_in,
                              void* d_out, int out_size, void* d_ws, size_t ws_size,
                              hipStream_t stream) {
    (void)in_sizes; (void)n_in; (void)out_size; (void)ws_size;

    const float* x   = (const float*)d_in[0];   // [N, C]
    const float* sim = (const float*)d_in[1];   // [C, E]
    const float* thr = (const float*)d_in[2];   // [1]
    const float* w1  = (const float*)d_in[3];   // [E, C, I]
    const float* w2  = (const float*)d_in[4];   // [E, I, C]

    float* outF    = (float*)d_out;
    float* finalO  = outF;                                     // N*C
    float* scoresO = outF + (size_t)NTOK * CH;                 // N*E
    int*   kO      = (int*)(outF + (size_t)NTOK * CH + (size_t)NTOK * NE);  // N (int32)

    // Workspace layout (bytes)
    char*   ws  = (char*)d_ws;
    __bf16* xb  = (__bf16*)(ws);                               // N*C*2      = 16 MiB
    __bf16* w1t = (__bf16*)(ws + (size_t)16777216);            // E*I*C*2    =  8 MiB
    __bf16* w2t = (__bf16*)(ws + (size_t)25165824);            // E*C*I*2    =  8 MiB
    float*  rw  = (float*) (ws + (size_t)33554432);            // N*E*4      = .5 MiB
    __bf16* Hb  = (__bf16*)(ws + (size_t)34078720);            // E*N*I*2    = 64 MiB

    // 1) Stage bf16 operands
    moe_conv_x <<<(NTOK * CH) / 256, 256, 0, stream>>>(x, xb);
    moe_conv_w1<<<(NE * IN * CH) / 256, 256, 0, stream>>>(w1, w1t);
    moe_conv_w2<<<(NE * CH * IN) / 256, 256, 0, stream>>>(w2, w2t);

    // 2) Gating (one wave per token; 8 tokens / 256-thread block)
    moe_gate<<<NTOK / 8, 256, 0, stream>>>(x, sim, thr, scoresO, kO, rw);

    // 3) GEMM1: 32 token groups x 4 col groups x 16 experts = 2048 blocks
    moe_gemm1<<<2048, 256, 0, stream>>>(xb, w1t, rw, Hb);

    // 4) GEMM2: 32 token groups x 16 col groups = 512 blocks
    moe_gemm2<<<512, 256, 0, stream>>>(Hb, w2t, finalO);
}
